// MultiheadAttention_12824772345984
// MI455X (gfx1250) — matrix operand order repaired and checked by reading, not compiled, not run on hardware
//
#include <hip/hip_runtime.h>

// ---------------- problem constants (match reference) ----------------
#define TT 4096   // frames
#define DD 1024   // feature dim
#define HH 8      // global heads
#define SS 4      // segments
#define EE 128    // D/H
#define LEE 256   // D/S
#define SEGL 1024 // T/S

// ---------------- types for WMMA / TDM ----------------
typedef __attribute__((ext_vector_type(4)))  unsigned int u32x4;
typedef __attribute__((ext_vector_type(4)))  int          i32x4;
typedef __attribute__((ext_vector_type(8)))  int          i32x8;
typedef __attribute__((ext_vector_type(16))) __bf16       bf16x16;
typedef __attribute__((ext_vector_type(8)))  float        f32x8;

union Frag { bf16x16 v; u32x4 q[2]; };

// Tensor Data Mover availability (device pass only; arity differs per toolchain)
#if defined(__HIP_DEVICE_COMPILE__) && __has_builtin(__builtin_amdgcn_tensor_load_to_lds) && __has_builtin(__builtin_amdgcn_s_wait_tensorcnt)
#define USE_TDM 1
#else
#define USE_TDM 0
#endif

__device__ __forceinline__ unsigned short f32_to_bf16(float f) {
  union { float f; unsigned int u; } x; x.f = f;
  unsigned int r = x.u + 0x7FFFu + ((x.u >> 16) & 1u);   // round-to-nearest-even
  return (unsigned short)(r >> 16);
}

#if USE_TDM
// Issue one TDM 2-D tile load: tile_dim0=64 halves (128B rows), tile_dim1=128 rows,
// LDS pad of 4 DWORDs (16B) after every 32 DWORDs (128B) -> row stride 144B = LDT halves.
__device__ __forceinline__ void tdm_load_tile64x128(unsigned lds_byte_addr,
                                                    const unsigned short* gptr,
                                                    unsigned rem_d0, unsigned rem_d1,
                                                    unsigned stride_d0_elems) {
  unsigned long long ga = (unsigned long long)(size_t)gptr;
  u32x4 g0;
  g0[0] = 1u;                                        // count=1 (valid), no gather
  g0[1] = lds_byte_addr;                             // lds_addr
  g0[2] = (unsigned)(ga & 0xFFFFFFFFu);              // global_addr[31:0]
  g0[3] = (unsigned)((ga >> 32) & 0x01FFFFFFu)       // global_addr[56:32]
          | (2u << 30);                              // type=2 ("image")
  i32x8 g1;
  // data_size=1 (2B) | pad_enable | pad_interval=4 (32 DW = 128B) | pad_amount=3 (4 DW = 16B)
  g1[0] = (int)((1u << 16) | (1u << 20) | (4u << 22) | (3u << 25));
  g1[1] = (int)((rem_d0 & 0xFFFFu) << 16);                               // tensor_dim0[15:0]
  g1[2] = (int)(((rem_d0 >> 16) & 0xFFFFu) | ((rem_d1 & 0xFFFFu) << 16)); // dim0[31:16]|dim1[15:0]
  g1[3] = (int)(((rem_d1 >> 16) & 0xFFFFu) | (64u << 16));               // dim1[31:16]|tile_dim0=64
  g1[4] = (int)(128u);                                                   // tile_dim1=128, tile_dim2=0
  g1[5] = (int)stride_d0_elems;                                          // dim0_stride[31:0]
  g1[6] = 0;                                                             // dim0_stride[47:32]|dim1_stride lo
  g1[7] = 0;
  i32x4 z4 = {0, 0, 0, 0};
#if defined(__clang_major__) && __clang_major__ >= 23
  i32x8 z8 = {0, 0, 0, 0, 0, 0, 0, 0};
  __builtin_amdgcn_tensor_load_to_lds(g0, g1, z4, z4, z8, 0);
#else
  __builtin_amdgcn_tensor_load_to_lds(g0, g1, z4, z4, 0);
#endif
}
#endif

// ---------------- f32 -> bf16 cast ----------------
__global__ void cast_f32_to_bf16(const float* __restrict__ src,
                                 unsigned short* __restrict__ dst, int n) {
  int i = blockIdx.x * blockDim.x + threadIdx.x;
  int stride = gridDim.x * blockDim.x;
  for (; i < n; i += stride) dst[i] = f32_to_bf16(src[i]);
}

// ---------------- bf16 WMMA GEMM ----------------
// C[M,N](f32 and/or bf16) = A[M,K] * B  ;  BT=false: B[K,N] row-major
//                                          BT=true : B[N,K] row-major (A @ B^T)
// All of M,N multiples of 128; K multiple of 64.  batch via blockIdx.z.
template <bool BT>
__global__ __launch_bounds__(256)
void gemm_bf16(const unsigned short* __restrict__ A,
               const unsigned short* __restrict__ B,
               float* __restrict__ Cf, unsigned short* __restrict__ Cb,
               int M, int N, int K, int lda, int ldb, int ldc,
               long sA, long sB, long sC) {
  constexpr int BM = 128, BN = 128, BK = 64, LDT = BK + 8; // 72 halves = 144B row (16B aligned)
  __shared__ unsigned short As[BM * LDT];
  __shared__ unsigned short Bs[BN * LDT];

  const int tid  = threadIdx.x;
  const int lane = tid & 31;
  const int wave = tid >> 5;          // 8 waves
  A += (long)blockIdx.z * sA;
  B += (long)blockIdx.z * sB;
  const long cBase = (long)blockIdx.z * sC;

  const int m0 = blockIdx.y * BM;
  const int n0 = blockIdx.x * BN;
  const int rowBase = (wave >> 1) * 32;  // 4 row groups of 32
  const int colBase = (wave & 1) * 64;   // 2 col groups of 64
  const int lr = lane & 15;              // row/col within 16-tile
  const int hs = lane >> 4;              // half-select per ISA layout

  f32x8 acc[2][4] = {};

  const int nK = K / BK;
  for (int ks = 0; ks < nK; ++ks) {
    const int k0 = ks * BK;

#if USE_TDM
    // ---- Tensor Data Mover: DMA the strided tile(s) straight into LDS ----
    if (wave == 0) {
      tdm_load_tile64x128((unsigned)(size_t)&As[0],
                          &A[(size_t)m0 * lda + k0],
                          (unsigned)(K - k0), (unsigned)(M - m0), (unsigned)lda);
      if (BT)
        tdm_load_tile64x128((unsigned)(size_t)&Bs[0],
                            &B[(size_t)n0 * ldb + k0],
                            (unsigned)(K - k0), (unsigned)(N - n0), (unsigned)ldb);
      __builtin_amdgcn_s_wait_tensorcnt(0);
    }
#else
    // ---- manual staging: batch loads into temps, then store (pipelined) ----
    {
      u32x4 ta[4];
#pragma unroll
      for (int it = 0; it < 4; ++it) {
        int row = (tid >> 3) + it * 32;
        int col = (tid & 7) * 8;
        ta[it] = *(const u32x4*)&A[(size_t)(m0 + row) * lda + k0 + col];
      }
#pragma unroll
      for (int it = 0; it < 4; ++it) {
        int row = (tid >> 3) + it * 32;
        int col = (tid & 7) * 8;
        *(u32x4*)&As[row * LDT + col] = ta[it];
      }
    }
    if (BT) {
      u32x4 tb[4];
#pragma unroll
      for (int it = 0; it < 4; ++it) {
        int row = (tid >> 3) + it * 32;
        int col = (tid & 7) * 8;
        tb[it] = *(const u32x4*)&B[(size_t)(n0 + row) * ldb + k0 + col];
      }
#pragma unroll
      for (int it = 0; it < 4; ++it) {
        int row = (tid >> 3) + it * 32;
        int col = (tid & 7) * 8;
        *(u32x4*)&Bs[row * LDT + col] = tb[it];
      }
    }
#endif
    if (!BT) {
      // transpose-stage B[K,N] -> Bs[n][k] (TDM cannot transpose)
      u32x4 tb[4];
#pragma unroll
      for (int it = 0; it < 4; ++it) {
        int idx = tid + it * 256;
        int k   = idx >> 4;           // 0..63
        int n   = (idx & 15) * 8;     // 0..120
        tb[it]  = *(const u32x4*)&B[(size_t)(k0 + k) * ldb + n0 + n];
      }
#pragma unroll
      for (int it = 0; it < 4; ++it) {
        int idx = tid + it * 256;
        int k   = idx >> 4;
        int n   = (idx & 15) * 8;
        const unsigned short* ds = (const unsigned short*)&tb[it];
#pragma unroll
        for (int i = 0; i < 8; ++i) Bs[(n + i) * LDT + k] = ds[i];
      }
    }
    // gfx1250 prefetch of next K tile (global_prefetch_b8)
    if (ks + 1 < nK) {
      __builtin_prefetch(&A[(size_t)(m0 + (tid >> 3)) * lda + k0 + BK], 0, 1);
      if (BT)
        __builtin_prefetch(&B[(size_t)(n0 + (tid >> 3)) * ldb + k0 + BK], 0, 1);
      else
        __builtin_prefetch(&B[(size_t)(k0 + BK + (tid >> 4)) * ldb + n0], 0, 1);
    }
    __syncthreads();

    // ---- 2 WMMA K-steps of 32 per staged tile ----
#pragma unroll
    for (int kk = 0; kk < BK; kk += 32) {
      Frag fa[2], fb[4];
#pragma unroll
      for (int mr = 0; mr < 2; ++mr) {
        // 16-bit A 16x32: lanes 0-15 K={0..7,16..23}, lanes 16-31 K={8..15,24..31}
        const unsigned short* p = &As[(rowBase + mr * 16 + lr) * LDT + kk + hs * 8];
        fa[mr].q[0] = *(const u32x4*)p;
        fa[mr].q[1] = *(const u32x4*)(p + 16);
      }
#pragma unroll
      for (int nc = 0; nc < 4; ++nc) {
        // 16-bit B 32x16: lanes 0-15 K={0..7,16..23}, lanes 16-31 K={8..15,24..31} (Bs is [n][k])
        const unsigned short* p = &Bs[(colBase + nc * 16 + lr) * LDT + kk + hs * 8];
        fb[nc].q[0] = *(const u32x4*)p;
        fb[nc].q[1] = *(const u32x4*)(p + 16);
      }
#pragma unroll
      for (int mr = 0; mr < 2; ++mr)
#pragma unroll
        for (int nc = 0; nc < 4; ++nc)
          acc[mr][nc] = __builtin_amdgcn_wmma_f32_16x16x32_bf16(
              false, fa[mr].v, false, fb[nc].v, (short)0, acc[mr][nc], false, false);
    }
    __syncthreads();
  }

  // ---- epilogue: C/D 16x16 layout: VGPR j -> M=j (lanes 0-15) / j+8 (16-31), N=lane%16
#pragma unroll
  for (int mr = 0; mr < 2; ++mr)
#pragma unroll
    for (int nc = 0; nc < 4; ++nc)
#pragma unroll
      for (int j = 0; j < 8; ++j) {
        int row = m0 + rowBase + mr * 16 + hs * 8 + j;
        int col = n0 + colBase + nc * 16 + lr;
        float v = acc[mr][nc][j];
        size_t off = (size_t)cBase + (size_t)row * ldc + col;
        if (Cf) Cf[off] = v;
        if (Cb) Cb[off] = f32_to_bf16(v);
      }
}

// ---------------- row softmax (no scale, per reference) ----------------
__global__ __launch_bounds__(256)
void softmax_rows(const float* __restrict__ S, float* __restrict__ outf,
                  unsigned short* __restrict__ outb, int cols) {
  const int row = blockIdx.x;
  const float* src = S + (size_t)row * cols;
  const int tid = threadIdx.x;
  __shared__ float red[8];

  float m = -3.4e38f;
  for (int c = tid; c < cols; c += 256) m = fmaxf(m, src[c]);
#pragma unroll
  for (int o = 16; o > 0; o >>= 1) m = fmaxf(m, __shfl_xor(m, o, 32));
  if ((tid & 31) == 0) red[tid >> 5] = m;
  __syncthreads();
  float mm = red[0];
#pragma unroll
  for (int i = 1; i < 8; ++i) mm = fmaxf(mm, red[i]);
  __syncthreads();

  float s = 0.f;
  for (int c = tid; c < cols; c += 256) s += __expf(src[c] - mm);
#pragma unroll
  for (int o = 16; o > 0; o >>= 1) s += __shfl_xor(s, o, 32);
  if ((tid & 31) == 0) red[tid >> 5] = s;
  __syncthreads();
  float ss = 0.f;
#pragma unroll
  for (int i = 0; i < 8; ++i) ss += red[i];
  float inv = 1.0f / ss;

  for (int c = tid; c < cols; c += 256) {
    float p = __expf(src[c] - mm) * inv;
    if (outf) outf[(size_t)row * cols + c] = p;
    outb[(size_t)row * cols + c] = f32_to_bf16(p);
  }
}

// ---------------- L2-normalize(x) + L2-normalize(local), emit bf16 ----------------
__global__ __launch_bounds__(256)
void l2norm_fuse(const float* __restrict__ x, const float* __restrict__ loc,
                 unsigned short* __restrict__ xgb) {
  const int row = blockIdx.x;
  const float* xr = x + (size_t)row * DD;
  const float* lo = loc + (size_t)row * DD;
  const int tid = threadIdx.x;
  __shared__ float redx[8], redl[8];

  float sx = 0.f, sl = 0.f;
  for (int c = tid; c < DD; c += 256) {
    float a = xr[c], b = lo[c];
    sx += a * a; sl += b * b;
  }
#pragma unroll
  for (int o = 16; o > 0; o >>= 1) {
    sx += __shfl_xor(sx, o, 32);
    sl += __shfl_xor(sl, o, 32);
  }
  if ((tid & 31) == 0) { redx[tid >> 5] = sx; redl[tid >> 5] = sl; }
  __syncthreads();
  float tx = 0.f, tl = 0.f;
#pragma unroll
  for (int i = 0; i < 8; ++i) { tx += redx[i]; tl += redl[i]; }
  float ix = 1.0f / fmaxf(sqrtf(tx), 1e-12f);
  float il = 1.0f / fmaxf(sqrtf(tl), 1e-12f);

  for (int c = tid; c < DD; c += 256)
    xgb[(size_t)row * DD + c] = f32_to_bf16(xr[c] * ix + lo[c] * il);
}

// ---------------- host orchestration ----------------
extern "C" void kernel_launch(void* const* d_in, const int* in_sizes, int n_in,
                              void* d_out, int out_size, void* d_ws, size_t ws_size,
                              hipStream_t stream) {
  (void)in_sizes; (void)n_in; (void)out_size; (void)ws_size;
  const float* x   = (const float*)d_in[0];
  const float* Wq  = (const float*)d_in[1];
  const float* Wk  = (const float*)d_in[2];
  const float* Wv  = (const float*)d_in[3];
  const float* Wo  = (const float*)d_in[4];
  const float* lWq = (const float*)d_in[5];
  const float* lWk = (const float*)d_in[6];
  const float* lWv = (const float*)d_in[7];
  const float* lWo = (const float*)d_in[8];

  char* ws = (char*)d_ws;
  size_t off = 0;
  auto alloc_us = [&](size_t n) -> unsigned short* {
    unsigned short* p = (unsigned short*)(ws + off);
    off += (n * 2 + 255) & ~(size_t)255; return p;
  };
  auto alloc_f = [&](size_t n) -> float* {
    float* p = (float*)(ws + off);
    off += (n * 4 + 255) & ~(size_t)255; return p;
  };

  const int T = TT, D = DD, H = HH, S = SS, E = EE, LE = LEE, seg = SEGL;

  unsigned short* xb   = alloc_us((size_t)T * D);
  unsigned short* Wqb  = alloc_us((size_t)H * D * E);
  unsigned short* Wkb  = alloc_us((size_t)H * D * E);
  unsigned short* Wvb  = alloc_us((size_t)H * D * E);
  unsigned short* Wob  = alloc_us((size_t)D * D);
  unsigned short* lWqb = alloc_us((size_t)S * D * LE);
  unsigned short* lWkb = alloc_us((size_t)S * D * LE);
  unsigned short* lWvb = alloc_us((size_t)S * D * LE);
  unsigned short* lWob = alloc_us((size_t)S * LE * D);
  unsigned short* qlb  = alloc_us((size_t)S * seg * LE);
  unsigned short* klb  = alloc_us((size_t)S * seg * LE);
  unsigned short* vlb  = alloc_us((size_t)S * seg * LE);
  float*          attl = alloc_f((size_t)S * seg * seg);
  unsigned short* Plb  = alloc_us((size_t)S * seg * seg);
  unsigned short* ylb  = alloc_us((size_t)S * seg * LE);
  float*          locf = alloc_f((size_t)T * D);
  unsigned short* xgb  = alloc_us((size_t)T * D);
  unsigned short* Qb   = alloc_us((size_t)H * T * E);
  unsigned short* Kb   = alloc_us((size_t)H * T * E);
  unsigned short* Vb   = alloc_us((size_t)H * T * E);
  float*          scf  = alloc_f((size_t)T * T);
  unsigned short* Pgb  = alloc_us((size_t)T * T);
  unsigned short* Ycat = alloc_us((size_t)T * D);

  dim3 blk(256);
  auto cast = [&](const float* s, unsigned short* d, size_t n) {
    int blocks = (int)((n + 1023) / 1024);
    cast_f32_to_bf16<<<blocks, blk, 0, stream>>>(s, d, (int)n);
  };
  cast(x, xb, (size_t)T * D);
  cast(Wq, Wqb, (size_t)H * D * E);
  cast(Wk, Wkb, (size_t)H * D * E);
  cast(Wv, Wvb, (size_t)H * D * E);
  cast(Wo, Wob, (size_t)D * D);
  cast(lWq, lWqb, (size_t)S * D * LE);
  cast(lWk, lWkb, (size_t)S * D * LE);
  cast(lWv, lWvb, (size_t)S * D * LE);
  cast(lWo, lWob, (size_t)S * LE * D);

  // ---- local attention (batched over S segments) ----
  gemm_bf16<false><<<dim3(LE / 128, seg / 128, S), blk, 0, stream>>>(
      xb, lWqb, nullptr, qlb, seg, LE, D, D, LE, LE,
      (long)seg * D, (long)D * LE, (long)seg * LE);
  gemm_bf16<false><<<dim3(LE / 128, seg / 128, S), blk, 0, stream>>>(
      xb, lWkb, nullptr, klb, seg, LE, D, D, LE, LE,
      (long)seg * D, (long)D * LE, (long)seg * LE);
  gemm_bf16<false><<<dim3(LE / 128, seg / 128, S), blk, 0, stream>>>(
      xb, lWvb, nullptr, vlb, seg, LE, D, D, LE, LE,
      (long)seg * D, (long)D * LE, (long)seg * LE);
  // scores = q @ k^T
  gemm_bf16<true><<<dim3(seg / 128, seg / 128, S), blk, 0, stream>>>(
      qlb, klb, attl, nullptr, seg, seg, LE, LE, LE, seg,
      (long)seg * LE, (long)seg * LE, (long)seg * seg);
  softmax_rows<<<S * seg, blk, 0, stream>>>(attl, nullptr, Plb, seg);
  // y = P @ v
  gemm_bf16<false><<<dim3(LE / 128, seg / 128, S), blk, 0, stream>>>(
      Plb, vlb, nullptr, ylb, seg, LE, seg, seg, LE, LE,
      (long)seg * seg, (long)seg * LE, (long)seg * LE);
  // local = y @ lWo
  gemm_bf16<false><<<dim3(D / 128, seg / 128, S), blk, 0, stream>>>(
      ylb, lWob, locf, nullptr, seg, D, LE, LE, D, D,
      (long)seg * LE, (long)LE * D, (long)seg * D);

  // ---- fusion ----
  l2norm_fuse<<<T, blk, 0, stream>>>(x, locf, xgb);

  // ---- global QKV (batched over H heads, A batch-stride 0) ----
  gemm_bf16<false><<<dim3(E / 128, T / 128, H), blk, 0, stream>>>(
      xgb, Wqb, nullptr, Qb, T, E, D, D, E, E, 0L, (long)D * E, (long)T * E);
  gemm_bf16<false><<<dim3(E / 128, T / 128, H), blk, 0, stream>>>(
      xgb, Wkb, nullptr, Kb, T, E, D, D, E, E, 0L, (long)D * E, (long)T * E);
  gemm_bf16<false><<<dim3(E / 128, T / 128, H), blk, 0, stream>>>(
      xgb, Wvb, nullptr, Vb, T, E, D, D, E, E, 0L, (long)D * E, (long)T * E);

  // ---- per-head: scores -> softmax -> AV (written into concat slot) ----
  for (int h = 0; h < H; ++h) {
    gemm_bf16<true><<<dim3(T / 128, T / 128, 1), blk, 0, stream>>>(
        Qb + (size_t)h * T * E, Kb + (size_t)h * T * E, scf, nullptr,
        T, T, E, E, E, T, 0L, 0L, 0L);
    float* attOut = (h == H - 1) ? ((float*)d_out + (size_t)T * D) : nullptr;
    softmax_rows<<<T, blk, 0, stream>>>(scf, attOut, Pgb, T);
    gemm_bf16<false><<<dim3(E / 128, T / 128, 1), blk, 0, stream>>>(
        Pgb, Vb + (size_t)h * T * E, nullptr, Ycat + (size_t)h * E,
        T, E, T, T, E, D, 0L, 0L, 0L);
  }

  // ---- output projection straight to d_out ----
  gemm_bf16<false><<<dim3(D / 128, T / 128, 1), blk, 0, stream>>>(
      Ycat, Wob, (float*)d_out, nullptr, T, D, D, D, D, D, 0L, 0L, 0L);
}
